// Attention_65678639891154
// MI455X (gfx1250) — matrix operand order repaired and checked by reading, not compiled, not run on hardware
//
#include <hip/hip_runtime.h>

// MI455X / gfx1250 attention: scores=(QK^T+pos)/sqrt(2D), mask, softmax, PV.
// bf16 WMMA for both GEMMs; raw scores staged through the p_attn output
// region (L2-resident); K/V tiles cooperatively staged in LDS (bf16, double
// buffered) so fragments come from ds_load_b128 instead of scalar VMEM.

typedef __attribute__((ext_vector_type(16))) __bf16 v16bf;
typedef __attribute__((ext_vector_type(2)))  __bf16 v2bf;
typedef __attribute__((ext_vector_type(8)))  float  v8f;
typedef __attribute__((ext_vector_type(4)))  float  vf4;   // clang vector: OK for NT builtins
typedef __attribute__((ext_vector_type(8)))  unsigned int v8u;

#define A_B   2
#define A_H   8
#define A_S   2048
#define A_D   64
#define KT    (A_S / 16)   // 128 key tiles (pass 1)
#define CH    (A_S / 32)   // 64 key chunks (pass 3)
#define NWAVE 4
#define ROWS_PER_BLOCK 128
#define KSTR  36           // K LDS row stride (dwords) = 72 bf16, 16B aligned, bank-safe
#define VSTR  20           // V LDS row stride (dwords) = 40 bf16, 16B aligned, bank-safe

__device__ __forceinline__ unsigned int pk_bf16(float a, float b) {
    v2bf t; t[0] = (__bf16)a; t[1] = (__bf16)b;
    return __builtin_bit_cast(unsigned int, t);
}

__device__ __forceinline__ v16bf frag_from_lds(const unsigned int* p) {
    uint4 lo = *reinterpret_cast<const uint4*>(p);      // ds_load_b128
    uint4 hi = *reinterpret_cast<const uint4*>(p + 4);  // ds_load_b128
    v8u u;
    u[0] = lo.x; u[1] = lo.y; u[2] = lo.z; u[3] = lo.w;
    u[4] = hi.x; u[5] = hi.y; u[6] = hi.z; u[7] = hi.w;
    return __builtin_bit_cast(v16bf, u);
}

__device__ __forceinline__ void load8cvt_s(const float* __restrict__ p, v16bf& d,
                                           const int lo, const float s) {
    float4 a = *reinterpret_cast<const float4*>(p);
    float4 b = *reinterpret_cast<const float4*>(p + 4);
    d[lo + 0] = (__bf16)(a.x * s); d[lo + 1] = (__bf16)(a.y * s);
    d[lo + 2] = (__bf16)(a.z * s); d[lo + 3] = (__bf16)(a.w * s);
    d[lo + 4] = (__bf16)(b.x * s); d[lo + 5] = (__bf16)(b.y * s);
    d[lo + 6] = (__bf16)(b.z * s); d[lo + 7] = (__bf16)(b.w * s);
}

__global__ __launch_bounds__(ROWS_PER_BLOCK, 1)
void attn_wmma_kernel(const float* __restrict__ Q, const float* __restrict__ K,
                      const float* __restrict__ V, const float* __restrict__ POS,
                      const int* __restrict__ MASK,
                      float* __restrict__ OUT, float* __restrict__ PATTN)
{
    __shared__ unsigned int ldsK[2][16 * KSTR];   // bf16 K tile, double buffered
    __shared__ unsigned int ldsV[2][64 * VSTR];   // bf16 V chunk (transposed [d][k])
    __shared__ float m_s[NWAVE][2][16];           // per-row softmax maxes

    const int tid  = threadIdx.x;
    const int lane = tid & 31;
    const int wv   = tid >> 5;
    const int n16  = lane & 15;
    const int half = lane >> 4;

    const int bi   = blockIdx.x;
    const int bh   = bi >> 4;
    const int qblk = bi & 15;
    const int b    = bh >> 3;
    const int qb   = qblk * ROWS_PER_BLOCK + wv * 32;

    const long base_qk = (long)bh * A_S * A_D;
    const long base_p  = (long)bh * A_S * A_S;
    const float scale  = 0.08838834764831845f;    // 1/sqrt(2*D)

    // ---- Q fragments (A layout), pre-scaled so epilogue is a single FMA ----
    v16bf aq[2][2];
    #pragma unroll
    for (int u = 0; u < 2; ++u) {
        const float* qp = Q + base_qk + (long)(qb + u * 16 + n16) * A_D;
        #pragma unroll
        for (int c = 0; c < 2; ++c) {
            const float* p0 = qp + c * 32 + half * 16;  // A: K = half*16+i
            load8cvt_s(p0,      aq[u][c], 0, scale);
            load8cvt_s(p0 + 8,  aq[u][c], 8, scale);
        }
    }

    float m[2][8];
    #pragma unroll
    for (int u = 0; u < 2; ++u)
        #pragma unroll
        for (int r = 0; r < 8; ++r) m[u][r] = -3.0e38f;

    // ---- Pass 1: stage K tiles in LDS (double buffered), QK^T via WMMA -----
    {   // stage tile 0
        const int krow = tid >> 3, dq = tid & 7;
        const float* gp = K + base_qk + (long)krow * A_D + dq * 8;
        float4 a = *reinterpret_cast<const float4*>(gp);
        float4 c = *reinterpret_cast<const float4*>(gp + 4);
        uint4 w;
        w.x = pk_bf16(a.x, a.y); w.y = pk_bf16(a.z, a.w);
        w.z = pk_bf16(c.x, c.y); w.w = pk_bf16(c.z, c.w);
        *reinterpret_cast<uint4*>(&ldsK[0][krow * KSTR + dq * 4]) = w;
    }
    __syncthreads();

    for (int j = 0; j < KT; ++j) {
        const int t = j & 1;
        if (j + 1 < KT) {   // stage-ahead next tile
            const int krow = tid >> 3, dq = tid & 7;
            const float* gp = K + base_qk + (long)((j + 1) * 16 + krow) * A_D + dq * 8;
            float4 a = *reinterpret_cast<const float4*>(gp);
            float4 c = *reinterpret_cast<const float4*>(gp + 4);
            uint4 w;
            w.x = pk_bf16(a.x, a.y); w.y = pk_bf16(a.z, a.w);
            w.z = pk_bf16(c.x, c.y); w.w = pk_bf16(c.z, c.w);
            *reinterpret_cast<uint4*>(&ldsK[t ^ 1][krow * KSTR + dq * 4]) = w;
        }

        const int col = j * 16 + n16;
        const int ms  = MASK[b * A_S + col];
        // B layout: lane = key column n16, element i -> dim = c*32 + half*16 + i
        v16bf bk0 = frag_from_lds(&ldsK[t][n16 * KSTR + half * 8]);
        v16bf bk1 = frag_from_lds(&ldsK[t][n16 * KSTR + 16 + half * 8]);

        if (j + 1 < KT)  // help the pos_attn HBM stream
            __builtin_prefetch(POS + base_p + (long)(qb + half * 8) * A_S + col + 16, 0, 1);

        #pragma unroll
        for (int u = 0; u < 2; ++u) {
            v8f acc = {};
            acc = __builtin_amdgcn_wmma_f32_16x16x32_bf16(false, aq[u][0], false, bk0,
                                                          (short)0, acc, false, false);
            acc = __builtin_amdgcn_wmma_f32_16x16x32_bf16(false, aq[u][1], false, bk1,
                                                          (short)0, acc, false, false);
            #pragma unroll
            for (int r = 0; r < 8; ++r) {   // C layout: row = r + 8*half, col = n16
                const int  row = qb + u * 16 + r + half * 8;
                const long off = base_p + (long)row * A_S + col;
                float pv = __builtin_nontemporal_load(&POS[off]);  // read-once stream
                float sc = fmaf(pv, scale, acc[r]);
                sc = (ms == 0) ? -1.0e9f : sc;
                PATTN[off] = sc;            // raw score staging: keep in L2 (regular)
                m[u][r] = fmaxf(m[u][r], sc);
            }
        }
        __syncthreads();
    }

    // ---- Row-max reduction within 16-lane halves, publish via LDS ----------
    #pragma unroll
    for (int u = 0; u < 2; ++u)
        #pragma unroll
        for (int r = 0; r < 8; ++r) {
            float x = m[u][r];
            x = fmaxf(x, __shfl_xor(x, 1, 32));
            x = fmaxf(x, __shfl_xor(x, 2, 32));
            x = fmaxf(x, __shfl_xor(x, 4, 32));
            x = fmaxf(x, __shfl_xor(x, 8, 32));
            m[u][r] = x;
        }
    if (n16 == 0) {
        #pragma unroll
        for (int u = 0; u < 2; ++u)
            #pragma unroll
            for (int r = 0; r < 8; ++r) m_s[wv][u][half * 8 + r] = m[u][r];
    }
    __syncthreads();
    float mrow[2], rl[2];
    #pragma unroll
    for (int u = 0; u < 2; ++u) mrow[u] = m_s[wv][u][n16];

    // ---- Pass 2: row sums of exp(s - m) (raw scores re-read from L2) -------
    #pragma unroll
    for (int u = 0; u < 2; ++u) {
        const int row = qb + u * 16 + n16;
        const float4* rp = reinterpret_cast<const float4*>(
            PATTN + base_p + (long)row * A_S + half * (A_S / 2));
        float lsum = 0.f;
        #pragma unroll 4
        for (int tt = 0; tt < (A_S / 2) / 4; ++tt) {
            float4 x = rp[tt];
            lsum += __expf(x.x - mrow[u]) + __expf(x.y - mrow[u]) +
                    __expf(x.z - mrow[u]) + __expf(x.w - mrow[u]);
        }
        lsum += __shfl_xor(lsum, 16, 32);
        rl[u] = 1.0f / lsum;
    }

    // ---- Pass 3: stage V in LDS (transposed), normalize+write p, PV WMMA ---
    v8f o[2][4] = {};
    {   // stage chunk 0: LDS layout [dim][key-pair packed bf16]
        const int kp = tid & 15, dg = tid >> 4;
        const float* g0 = V + base_qk + (long)(2 * kp) * A_D + dg * 8;
        float4 a0 = *reinterpret_cast<const float4*>(g0);
        float4 b0 = *reinterpret_cast<const float4*>(g0 + 4);
        float4 a1 = *reinterpret_cast<const float4*>(g0 + A_D);
        float4 b1 = *reinterpret_cast<const float4*>(g0 + A_D + 4);
        float f0[8] = {a0.x, a0.y, a0.z, a0.w, b0.x, b0.y, b0.z, b0.w};
        float f1[8] = {a1.x, a1.y, a1.z, a1.w, b1.x, b1.y, b1.z, b1.w};
        #pragma unroll
        for (int e = 0; e < 8; ++e)
            ldsV[0][(dg * 8 + e) * VSTR + kp] = pk_bf16(f0[e], f1[e]);
    }
    __syncthreads();

    for (int c = 0; c < CH; ++c) {
        const int t = c & 1;
        if (c + 1 < CH) {   // stage-ahead next V chunk
            const int kp = tid & 15, dg = tid >> 4;
            const float* g0 = V + base_qk + (long)((c + 1) * 32 + 2 * kp) * A_D + dg * 8;
            float4 a0 = *reinterpret_cast<const float4*>(g0);
            float4 b0 = *reinterpret_cast<const float4*>(g0 + 4);
            float4 a1 = *reinterpret_cast<const float4*>(g0 + A_D);
            float4 b1 = *reinterpret_cast<const float4*>(g0 + A_D + 4);
            float f0[8] = {a0.x, a0.y, a0.z, a0.w, b0.x, b0.y, b0.z, b0.w};
            float f1[8] = {a1.x, a1.y, a1.z, a1.w, b1.x, b1.y, b1.z, b1.w};
            #pragma unroll
            for (int e = 0; e < 8; ++e)
                ldsV[t ^ 1][(dg * 8 + e) * VSTR + kp] = pk_bf16(f0[e], f1[e]);
        }

        // V B-fragments from LDS: lane = dim column, element i -> key = half*16 + i
        v16bf bv[4];
        #pragma unroll
        for (int nt = 0; nt < 4; ++nt)
            bv[nt] = frag_from_lds(&ldsV[t][(nt * 16 + n16) * VSTR + half * 8]);

        #pragma unroll
        for (int u = 0; u < 2; ++u) {
            const int row = qb + u * 16 + n16;     // A layout: row = n16
            float* rp = PATTN + base_p + (long)row * A_S + c * 32 + half * 16;
            float raw[16];
            *reinterpret_cast<float4*>(&raw[0])  = *reinterpret_cast<const float4*>(rp);
            *reinterpret_cast<float4*>(&raw[4])  = *reinterpret_cast<const float4*>(rp + 4);
            *reinterpret_cast<float4*>(&raw[8])  = *reinterpret_cast<const float4*>(rp + 8);
            *reinterpret_cast<float4*>(&raw[12]) = *reinterpret_cast<const float4*>(rp + 12);

            v16bf ap;
            #pragma unroll
            for (int i = 0; i < 16; ++i) {
                float pv = __expf(raw[i] - mrow[u]) * rl[u];
                ap[i]  = (__bf16)pv;
                raw[i] = pv;
            }
            // final normalized probabilities: write-once -> non-temporal b128 stores
            vf4 w0 = {raw[0],  raw[1],  raw[2],  raw[3]};
            vf4 w1 = {raw[4],  raw[5],  raw[6],  raw[7]};
            vf4 w2 = {raw[8],  raw[9],  raw[10], raw[11]};
            vf4 w3 = {raw[12], raw[13], raw[14], raw[15]};
            __builtin_nontemporal_store(w0, reinterpret_cast<vf4*>(rp));
            __builtin_nontemporal_store(w1, reinterpret_cast<vf4*>(rp + 4));
            __builtin_nontemporal_store(w2, reinterpret_cast<vf4*>(rp + 8));
            __builtin_nontemporal_store(w3, reinterpret_cast<vf4*>(rp + 12));

            #pragma unroll
            for (int nt = 0; nt < 4; ++nt)
                o[u][nt] = __builtin_amdgcn_wmma_f32_16x16x32_bf16(false, ap, false, bv[nt],
                                                                   (short)0, o[u][nt], false, false);
        }
        __syncthreads();
    }

    // ---- Store OUT (C layout, write-once -> non-temporal) ------------------
    #pragma unroll
    for (int u = 0; u < 2; ++u)
        #pragma unroll
        for (int nt = 0; nt < 4; ++nt)
            #pragma unroll
            for (int r = 0; r < 8; ++r) {
                const int row = qb + u * 16 + r + half * 8;
                __builtin_nontemporal_store(
                    o[u][nt][r], &OUT[base_qk + (long)row * A_D + nt * 16 + n16]);
            }
}

extern "C" void kernel_launch(void* const* d_in, const int* in_sizes, int n_in,
                              void* d_out, int out_size, void* d_ws, size_t ws_size,
                              hipStream_t stream) {
    (void)in_sizes; (void)n_in; (void)out_size; (void)d_ws; (void)ws_size;
    const float* q    = (const float*)d_in[0];
    const float* k    = (const float*)d_in[1];
    const float* v    = (const float*)d_in[2];
    const float* pos  = (const float*)d_in[3];
    const int*   mask = (const int*)d_in[4];

    float* out   = (float*)d_out;
    float* pattn = out + (long)A_B * A_H * A_S * A_D;   // outputs concatenated (out, p_attn)

    const int nblocks = (A_B * A_H * A_S) / ROWS_PER_BLOCK;  // 256
    attn_wmma_kernel<<<dim3(nblocks), dim3(ROWS_PER_BLOCK), 0, stream>>>(
        q, k, v, pos, mask, out, pattn);
}
